// GNNCompleteNeighborhood_55095840473790
// MI455X (gfx1250) — matrix operand order repaired and checked by reading, not compiled, not run on hardware
//
#include <hip/hip_runtime.h>
#include <math.h>

typedef _Float16 f16;
typedef __attribute__((ext_vector_type(16))) _Float16 v16h;
typedef __attribute__((ext_vector_type(8)))  _Float16 v8h;
typedef __attribute__((ext_vector_type(8)))  float    v8f;

#define N_ROWS 8192
#define D_IN   128
#define D_H    256
#define D_OUT  10
#define N_G    64

// ---------------- WMMA fragment loaders (wave32) ----------------
__device__ __forceinline__ v16h load_frag_A(const f16* base, int ld, int mblk, int k0) {
    int lane = threadIdx.x & 31;
    int row  = mblk + (lane & 15);
    int koff = k0 + ((lane >> 4) ? 16 : 0);
    const f16* p = base + (size_t)row * ld + koff;
    v8h lo = *(const v8h*)p;
    v8h hi = *(const v8h*)(p + 8);
    v16h a;
#pragma unroll
    for (int i = 0; i < 8; ++i) { a[i] = lo[i]; a[i + 8] = hi[i]; }
    return a;
}

__device__ __forceinline__ v16h load_frag_B(const f16* baseT, int ld, int nblk, int k0) {
    int lane = threadIdx.x & 31;
    int col  = nblk + (lane & 15);
    int koff = k0 + ((lane >> 4) ? 16 : 0);
    return *(const v16h*)(baseT + (size_t)col * ld + koff);
}

// ---------------- K1: prep -------------------------------------------------
__global__ void prep_kernel(const float* __restrict__ x, const int* __restrict__ batch,
                            const float* __restrict__ W1,
                            f16* __restrict__ x16, f16* __restrict__ xn_rm,
                            f16* __restrict__ xnT, float* __restrict__ cdiag,
                            f16* __restrict__ W1T, float* __restrict__ counts) {
    if (blockIdx.x == gridDim.x - 1) {
        int n = threadIdx.x;
        if (n < D_H)
            for (int k = 0; k < D_IN; ++k)
                W1T[(size_t)n * D_IN + k] = (f16)W1[(size_t)k * D_H + n];
        return;
    }
    int i = blockIdx.x * blockDim.x + threadIdx.x;
    if (i >= N_ROWS) return;
    float ss = 0.f;
    for (int k = 0; k < D_IN; ++k) { float v = x[(size_t)i * D_IN + k]; ss += v * v; }
    float inv = 1.0f / fmaxf(sqrtf(ss), 1e-8f);
    float snn = 0.f;
    for (int k = 0; k < D_IN; ++k) {
        float v  = x[(size_t)i * D_IN + k];
        f16 xnv  = (f16)(v * inv);
        x16[(size_t)i * D_IN + k]   = (f16)v;
        xn_rm[(size_t)i * D_IN + k] = xnv;
        xnT[(size_t)k * N_ROWS + i] = xnv;
        float xf = (float)xnv;
        snn += xf * xf;
    }
    cdiag[i] = 1.0f - snn;
    atomicAdd(&counts[batch[i]], 1.0f);
}

// ---------------- K2: h = relu(x@W1 + b1) ---------------------------------
__global__ void fc1_kernel(const f16* __restrict__ x16, const f16* __restrict__ W1T,
                           const float* __restrict__ b1,
                           f16* __restrict__ hT, f16* __restrict__ h_rm) {
    int w = threadIdx.x >> 5, lane = threadIdx.x & 31;
    int mblk = (blockIdx.x * 4 + w) * 16;
    int nblk = blockIdx.y * 16;
    v8f acc = {};
#pragma unroll
    for (int k0 = 0; k0 < D_IN; k0 += 32) {
        v16h a = load_frag_A(x16, D_IN, mblk, k0);
        v16h b = load_frag_B(W1T, D_IN, nblk, k0);
        acc = __builtin_amdgcn_wmma_f32_16x16x32_f16(false, a, false, b, (short)0, acc, false, false);
    }
    int n     = nblk + (lane & 15);
    int mbase = mblk + ((lane >> 4) ? 8 : 0);
    float bias = b1[n];
#pragma unroll
    for (int r = 0; r < 8; ++r) {
        int i   = mbase + r;
        float v = acc[r] + bias;
        v = v > 0.f ? v : 0.f;
        f16 hv = (f16)v;
        h_rm[(size_t)i * D_H + n]  = hv;
        hT[(size_t)n * N_ROWS + i] = hv;
    }
}

// ---------------- K3: Mmat += xn^T @ h ------------------------------------
__global__ void xtx_kernel(const f16* __restrict__ xnT, const f16* __restrict__ hT,
                           float* __restrict__ Mmat) {
    int w = threadIdx.x >> 5, lane = threadIdx.x & 31;
    int mblk   = blockIdx.x * 16;
    int nblk   = blockIdx.y * 16;
    int kstart = (blockIdx.z * 4 + w) * 512;
    v8f acc = {};
    for (int k0 = kstart; k0 < kstart + 512; k0 += 32) {
        v16h a = load_frag_A(xnT, N_ROWS, mblk, k0);
        v16h b = load_frag_B(hT,  N_ROWS, nblk, k0);
        acc = __builtin_amdgcn_wmma_f32_16x16x32_f16(false, a, false, b, (short)0, acc, false, false);
    }
    int n     = nblk + (lane & 15);
    int mbase = mblk + ((lane >> 4) ? 8 : 0);
#pragma unroll
    for (int r = 0; r < 8; ++r)
        atomicAdd(&Mmat[(size_t)(mbase + r) * D_H + n], acc[r]);
}

// ---------------- K3b: Mmat -> MmatT f16 ----------------------------------
__global__ void mtr_kernel(const float* __restrict__ Mmat, f16* __restrict__ MmatT) {
    int idx = blockIdx.x * blockDim.x + threadIdx.x;
    if (idx >= D_IN * D_H) return;
    int m = idx / D_H, n = idx % D_H;
    MmatT[(size_t)n * D_IN + m] = (f16)Mmat[idx];
}

// ---------------- K4: agg = xn@M + c*h, pool ------------------------------
__global__ void agg_kernel(const f16* __restrict__ xn_rm, const f16* __restrict__ MmatT,
                           const f16* __restrict__ h_rm, const float* __restrict__ cdiag,
                           const int* __restrict__ batch, float* __restrict__ pooled) {
    int w = threadIdx.x >> 5, lane = threadIdx.x & 31;
    int mblk = (blockIdx.x * 4 + w) * 16;
    int nblk = blockIdx.y * 16;
    v8f acc = {};
#pragma unroll
    for (int k0 = 0; k0 < D_IN; k0 += 32) {
        v16h a = load_frag_A(xn_rm, D_IN, mblk, k0);
        v16h b = load_frag_B(MmatT, D_IN, nblk, k0);
        acc = __builtin_amdgcn_wmma_f32_16x16x32_f16(false, a, false, b, (short)0, acc, false, false);
    }
    int n     = nblk + (lane & 15);
    int mbase = mblk + ((lane >> 4) ? 8 : 0);
#pragma unroll
    for (int r = 0; r < 8; ++r) {
        int i   = mbase + r;
        float v = acc[r] + cdiag[i] * (float)h_rm[(size_t)i * D_H + n];
        atomicAdd(&pooled[(size_t)batch[i] * D_H + n], v);
    }
}

// ---------------- K5: head -------------------------------------------------
__global__ void head_kernel(const float* __restrict__ pooled, const float* __restrict__ counts,
                            const float* __restrict__ W2, const float* __restrict__ b2,
                            float* __restrict__ out) {
    int g = threadIdx.x;
    if (g >= N_G) return;
    float cnt = fmaxf(counts[g], 1.0f);
    float logits[D_OUT];
#pragma unroll
    for (int c = 0; c < D_OUT; ++c) logits[c] = b2[c];
    for (int n = 0; n < D_H; ++n) {
        float m = pooled[(size_t)g * D_H + n] / cnt;
#pragma unroll
        for (int c = 0; c < D_OUT; ++c) logits[c] += m * W2[(size_t)n * D_OUT + c];
    }
    float mx = logits[0];
#pragma unroll
    for (int c = 1; c < D_OUT; ++c) mx = fmaxf(mx, logits[c]);
    float s = 0.f;
#pragma unroll
    for (int c = 0; c < D_OUT; ++c) s += expf(logits[c] - mx);
    float lse = logf(s);
#pragma unroll
    for (int c = 0; c < D_OUT; ++c) out[(size_t)g * D_OUT + c] = logits[c] - mx - lse;
}

// ---------------------------- launcher ------------------------------------
extern "C" void kernel_launch(void* const* d_in, const int* in_sizes, int n_in,
                              void* d_out, int out_size, void* d_ws, size_t ws_size,
                              hipStream_t stream) {
    const float* x     = (const float*)d_in[0];
    const int*   batch = (const int*)d_in[1];
    const float* W1    = (const float*)d_in[2];
    const float* b1    = (const float*)d_in[3];
    const float* W2    = (const float*)d_in[4];
    const float* b2    = (const float*)d_in[5];
    float* out = (float*)d_out;
    (void)in_sizes; (void)n_in; (void)out_size; (void)ws_size;

    char* ws = (char*)d_ws;
    size_t off = 0;
    f16*   x16    = (f16*)(ws + off); off += (size_t)N_ROWS * D_IN * 2;
    f16*   xn_rm  = (f16*)(ws + off); off += (size_t)N_ROWS * D_IN * 2;
    f16*   xnT    = (f16*)(ws + off); off += (size_t)D_IN * N_ROWS * 2;
    f16*   hT     = (f16*)(ws + off); off += (size_t)D_H * N_ROWS * 2;
    f16*   h_rm   = (f16*)(ws + off); off += (size_t)N_ROWS * D_H * 2;
    f16*   W1T    = (f16*)(ws + off); off += (size_t)D_H * D_IN * 2;
    float* cdiag  = (float*)(ws + off); off += (size_t)N_ROWS * 4;
    float* Mmat   = (float*)(ws + off); off += (size_t)D_IN * D_H * 4;
    f16*   MmatT  = (f16*)(ws + off); off += (size_t)D_H * D_IN * 2;
    float* pooled = (float*)(ws + off); off += (size_t)N_G * D_H * 4;
    float* counts = (float*)(ws + off); off += (size_t)N_G * 4;

    hipMemsetAsync(Mmat,   0, (size_t)D_IN * D_H * 4, stream);
    hipMemsetAsync(pooled, 0, (size_t)N_G * D_H * 4, stream);
    hipMemsetAsync(counts, 0, (size_t)N_G * 4, stream);

    prep_kernel<<<dim3(N_ROWS / 256 + 1), dim3(256), 0, stream>>>(
        x, batch, W1, x16, xn_rm, xnT, cdiag, W1T, counts);

    fc1_kernel<<<dim3(N_ROWS / 64, D_H / 16), dim3(128), 0, stream>>>(
        x16, W1T, b1, hT, h_rm);

    xtx_kernel<<<dim3(D_IN / 16, D_H / 16, 4), dim3(128), 0, stream>>>(
        xnT, hT, Mmat);

    mtr_kernel<<<dim3((D_IN * D_H + 255) / 256), dim3(256), 0, stream>>>(Mmat, MmatT);

    agg_kernel<<<dim3(N_ROWS / 64, D_H / 16), dim3(128), 0, stream>>>(
        xn_rm, MmatT, h_rm, cdiag, batch, pooled);

    head_kernel<<<dim3(1), dim3(64), 0, stream>>>(pooled, counts, W2, b2, out);
}
